// SpatialLoss_75101798138307
// MI455X (gfx1250) — compile-verified, matrix operand order checked by reading, not run on hardware
//
#include <hip/hip_runtime.h>
#include <hip/hip_bf16.h>
#include <math.h>

// ---------------- problem constants ----------------
#define NUM_CLASS 5
#define B_   16
#define H_   512
#define W_   512
#define HW   (H_ * W_)            // 262144 = 2^18
#define NPIX (B_ * HW)            // 4194304
#define NWORDS (NPIX / 32)        // 131072 bit-packed words
#define WPB   8192                // words per batch plane = 512*16
#define CHW_PRED (NUM_CLASS * HW)
#define CHW_IMG  (3 * HW)
#define IGNORE_L 255
#define ENT_CAP  0.4f
#define PCT_FRAC 0.60f
#define SIM_THR  2.5f

// fused-dilation tiling: 5 iterations per launch, 5-row halo
#define TROWS 16
#define HALO  5
#define LROWS (TROWS + 2 * HALO)     // 26
#define TWORDS (LROWS * 16 * 16)     // 6656 words = 26 KB LDS

typedef __attribute__((ext_vector_type(2))) float v2f;
typedef __attribute__((ext_vector_type(8))) float v8f;

// device-resident scalars shared between dependent kernels (graph-safe, no host sync)
struct Scal {
  unsigned n;
  int      has;
  float    frac;
  float    thresh;
  unsigned binHi[2];
  unsigned rankRem[2];
  float    sval[2];
  float    proto;
};

// ---------------- helpers ----------------
__device__ __forceinline__ unsigned flipKey(float f) {
  unsigned u = __float_as_uint(f);
  return (u & 0x80000000u) ? ~u : (u | 0x80000000u);   // monotonic total order on floats
}
__device__ __forceinline__ float unflipKey(unsigned k) {
  unsigned u = (k & 0x80000000u) ? (k & 0x7FFFFFFFu) : ~k;
  return __uint_as_float(u);
}

// Deterministic 256-thread block reduction using the matrix pipe:
// wave 0 folds the 256 partials with A=ones via V_WMMA_F32_16X16X4_F32
// (f32 in, f32 accumulate -> exact FP32 adds in a fixed order).
__device__ __forceinline__ float block_reduce_256_wmma(float v, float* buf) {
  const int tid = threadIdx.x;
  buf[tid] = v;
  __syncthreads();
  float total = 0.0f;
  if (tid < 32) {                       // wave 0, EXEC all-ones (WMMA requirement)
    const int lane = tid;
    float s;
#if defined(__gfx1250__) && __has_builtin(__builtin_amdgcn_wmma_f32_16x16x4_f32)
    v8f acc = {};
    v2f ones; ones[0] = 1.0f; ones[1] = 1.0f;
#pragma unroll
    for (int k = 0; k < 4; ++k) {       // 4 x 64 = 256 values
      v2f bv; bv[0] = buf[k * 64 + lane * 2]; bv[1] = buf[k * 64 + lane * 2 + 1];
      acc = __builtin_amdgcn_wmma_f32_16x16x4_f32(false, ones, false, bv,
                                                  (short)0, acc, false, false);
    }
    s = acc[0];                         // row 0 of D: column sums
#elif defined(__gfx1250__)
    // fallback: codegen-confirmed f16 WMMA (exact for 0/1 counts)
    typedef __attribute__((ext_vector_type(16))) _Float16 v16h;
    v16h av, bv;
#pragma unroll
    for (int j = 0; j < 16; ++j) {
      av[j] = (_Float16)1.0f;
      int idx = lane * 16 + j;
      bv[j] = (_Float16)((idx < 256) ? buf[idx] : 0.0f);
    }
    v8f acc = {};
    acc = __builtin_amdgcn_wmma_f32_16x16x32_f16(false, av, false, bv,
                                                 (short)0, acc, false, false);
    s = acc[0];
#else
    // host-pass / generic parse path
    s = 0.0f;
    if (lane == 0) for (int j = 0; j < 256; ++j) s += buf[j];
#endif
    // sum the 16 column sums (butterfly stays within 16-lane groups)
    s += __shfl_xor(s, 1, 32);
    s += __shfl_xor(s, 2, 32);
    s += __shfl_xor(s, 4, 32);
    s += __shfl_xor(s, 8, 32);
    total = s;
  }
  __syncthreads();
  return total;  // valid on thread 0
}

// ---------------- kernels ----------------

// softmax / argmax / entropy / log-sum-exp (single pass over 84 MB of pred)
__global__ void k_softmax(const float* __restrict__ pred, float* __restrict__ ent,
                          float* __restrict__ lse, int* __restrict__ pseu) {
  const int p = blockIdx.x * blockDim.x + threadIdx.x;
  if (p >= NPIX) return;
  const int b = p >> 18, r = p & (HW - 1);
  const float* __restrict__ base = pred + (size_t)b * CHW_PRED + r;
  float x[NUM_CLASS];
#pragma unroll
  for (int c = 0; c < NUM_CLASS; ++c) x[c] = base[(size_t)c * HW];
  float m = x[0]; int am = 0;
#pragma unroll
  for (int c = 1; c < NUM_CLASS; ++c) if (x[c] > m) { m = x[c]; am = c; }  // first-max like jnp.argmax
  float e[NUM_CLASS], s = 0.f;
#pragma unroll
  for (int c = 0; c < NUM_CLASS; ++c) { e[c] = expf(x[c] - m); s += e[c]; }
  const float inv = 1.0f / s;
  float Hh = 0.f;
#pragma unroll
  for (int c = 0; c < NUM_CLASS; ++c) { float pr = e[c] * inv; Hh -= pr * logf(pr + 1e-10f); }
  ent[p]  = Hh;
  lse[p]  = m + logf(s);
  pseu[p] = am;
}

__global__ void k_clear(unsigned* hist, Scal* sc, int resetSc) {
  const int i = blockIdx.x * blockDim.x + threadIdx.x;
  if (i < 65536) hist[i] = 0u;
  if (resetSc && i == 0) {
    sc->n = 0; sc->has = 0; sc->frac = 0.f; sc->thresh = ENT_CAP;
    sc->binHi[0] = sc->binHi[1] = 0u;
    sc->rankRem[0] = sc->rankRem[1] = 0u;
    sc->sval[0] = sc->sval[1] = __builtin_inff();   // empty percentile -> inf (ref semantics)
    sc->proto = 0.f;
  }
}

// radix pass 1: histogram of high 16 bits of the flipped entropy key over mask_i
__global__ void k_hist_hi(const int* __restrict__ pseu, const float* __restrict__ ent,
                          unsigned* hist, int cls) {
  const int stride = gridDim.x * blockDim.x;
  for (int p = blockIdx.x * blockDim.x + threadIdx.x; p < NPIX; p += stride)
    if (pseu[p] == cls) atomicAdd(&hist[flipKey(ent[p]) >> 16], 1u);
}

// locate high-bin + residual rank for ranks lo = floor(0.6*(n-1)) and hi = min(lo+1, n-1)
__global__ void k_select_hi(const unsigned* __restrict__ hist, Scal* sc) {
  __shared__ unsigned tsum[256];
  __shared__ unsigned tpre[257];
  __shared__ unsigned rtgt[2];
  const int tid = threadIdx.x;
  const unsigned* hp = hist + tid * 256;
  unsigned s = 0;
  for (int j = 0; j < 256; ++j) s += hp[j];
  tsum[tid] = s;
  __syncthreads();
  if (tid == 0) {
    unsigned c = 0;
    for (int i = 0; i < 256; ++i) { tpre[i] = c; c += tsum[i]; }
    tpre[256] = c;
    const unsigned n = c;
    sc->n = n; sc->has = (n > 0) ? 1 : 0;
    unsigned lo = 0, hi = 0; float frac = 0.f;
    if (n > 0) {
      float q = PCT_FRAC * (float)(n - 1);   // f32, matches jnp weak-typed math
      lo = (unsigned)q; if (lo > n - 1) lo = n - 1;
      hi = lo + 1;      if (hi > n - 1) hi = n - 1;
      frac = q - (float)lo;
    }
    sc->frac = frac;
    rtgt[0] = lo; rtgt[1] = hi;
  }
  __syncthreads();
  const unsigned n = tpre[256];
  if (n == 0) return;
  for (int slot = 0; slot < 2; ++slot) {
    const unsigned r = rtgt[slot];
    if (r >= tpre[tid] && r < tpre[tid] + tsum[tid]) {
      unsigned rem = r - tpre[tid], c = 0;
      for (int j = 0; j < 256; ++j) {
        const unsigned h = hp[j];
        if (rem < c + h) { sc->binHi[slot] = (unsigned)(tid * 256 + j); sc->rankRem[slot] = rem - c; break; }
        c += h;
      }
    }
  }
}

// radix pass 2: low 16 bits within the selected high bin
__global__ void k_hist_lo(const int* __restrict__ pseu, const float* __restrict__ ent,
                          unsigned* hist, const Scal* __restrict__ sc, int cls, int slot) {
  const unsigned target = sc->binHi[slot];
  const int stride = gridDim.x * blockDim.x;
  for (int p = blockIdx.x * blockDim.x + threadIdx.x; p < NPIX; p += stride) {
    if (pseu[p] == cls) {
      const unsigned k = flipKey(ent[p]);
      if ((k >> 16) == target) atomicAdd(&hist[k & 0xFFFFu], 1u);
    }
  }
}

__global__ void k_select_lo(const unsigned* __restrict__ hist, Scal* sc, int slot) {
  __shared__ unsigned tsum[256];
  __shared__ unsigned tpre[257];
  const int tid = threadIdx.x;
  const unsigned* hp = hist + tid * 256;
  unsigned s = 0;
  for (int j = 0; j < 256; ++j) s += hp[j];
  tsum[tid] = s;
  __syncthreads();
  if (tid == 0) {
    unsigned c = 0;
    for (int i = 0; i < 256; ++i) { tpre[i] = c; c += tsum[i]; }
    tpre[256] = c;
  }
  __syncthreads();
  const unsigned total = tpre[256];
  if (total == 0) return;                 // sval stays +inf
  unsigned r = sc->rankRem[slot];
  if (r >= total) r = total - 1;
  if (r >= tpre[tid] && r < tpre[tid] + tsum[tid]) {
    unsigned rem = r - tpre[tid], c = 0;
    for (int j = 0; j < 256; ++j) {
      const unsigned h = hp[j];
      if (rem < c + h) {
        const unsigned key = (sc->binHi[slot] << 16) | (unsigned)(tid * 256 + j);
        sc->sval[slot] = unflipKey(key);
        break;
      }
      c += h;
    }
  }
}

__global__ void k_thresh(Scal* sc) {
  if (blockIdx.x == 0 && threadIdx.x == 0) {
    float t = ENT_CAP;
    if (sc->has) {
      const float p = sc->sval[0] * (1.0f - sc->frac) + sc->sval[1] * sc->frac;
      t = fminf(p, ENT_CAP);
    }
    sc->thresh = t;
  }
}

// threshold -> IGNORE, build bit-packed mask_i2, reduce masked img1 sum + count (WMMA reduce)
__global__ void k_apply(int* pseu, const float* __restrict__ ent, const float* __restrict__ img,
                        unsigned* __restrict__ maskA, const Scal* __restrict__ sc, int cls,
                        float* partialS, float* partialC) {
  __shared__ float buf[256];
  const int w = blockIdx.x * 256 + threadIdx.x;     // exact: NWORDS threads
  const float thresh = sc->thresh;
  const int pbase = w * 32;
  const int b = pbase >> 18, r0 = pbase & (HW - 1); // 32 px share batch & row
  const float* __restrict__ im1 = img + (size_t)b * CHW_IMG + HW + r0;
  unsigned bits = 0u; float msum = 0.f, mcnt = 0.f;
#pragma unroll 4
  for (int j = 0; j < 32; ++j) {
    const int p = pbase + j;
    if (pseu[p] == cls) {
      if (ent[p] >= thresh) pseu[p] = IGNORE_L;
      else { bits |= (1u << j); msum += im1[j]; mcnt += 1.f; }
    }
  }
  maskA[w] = bits;
  const float S = block_reduce_256_wmma(msum, buf);
  const float C = block_reduce_256_wmma(mcnt, buf);
  if (threadIdx.x == 0) { partialS[blockIdx.x] = S; partialC[blockIdx.x] = C; }
}

__global__ void k_proto(const float* __restrict__ partialS, const float* __restrict__ partialC,
                        Scal* sc, int nparts) {
  __shared__ float buf[256];
  float s = 0.f, c = 0.f;
  for (int i = threadIdx.x; i < nparts; i += 256) { s += partialS[i]; c += partialC[i]; }
  const float S = block_reduce_256_wmma(s, buf);
  const float C = block_reduce_256_wmma(c, buf);
  if (threadIdx.x == 0) {
    long cnt = (long)(C + 0.5f);
    if (cnt < 1) cnt = 1;
    sc->proto = S / (float)cnt;
  }
}

// Five 3-D connectivity-1 cross dilation iterations per launch (time-tiled in LDS).
// Tile: all 16 batches x 26 rows (16 center + 5 halo each side) x 16 words.
// Batch & width dims are fully resident, so only the row dim shrinks per iteration;
// after 5 iterations the 16 center rows are exact. Tile staging uses the gfx1250
// async LDS data movers (ASYNCcnt) with a portable fallback.
__global__ void k_dilate5(const unsigned* __restrict__ src, unsigned* __restrict__ dst) {
  __shared__ unsigned tile[TWORDS];
  const int tid = threadIdx.x;           // 256 threads: fixed (b, w) per thread
  const int h0  = blockIdx.x * TROWS;    // 32 blocks cover 512 rows
  const int w   = tid & 15;
  const int b   = (tid >> 4) & 15;
  unsigned cur[LROWS];

  // stage tile rows [h0-5, h0+21) into LDS (zeros outside the image)
#pragma unroll
  for (int k = 0; k < LROWS; ++k) {
    const int h = h0 - HALO + k;
    const int l = tid + 256 * k;
    if (h >= 0 && h < H_) {
      const unsigned goff = (unsigned)((b * WPB + h * 16 + w) * 4);
#if defined(__gfx1250__)
      const unsigned ldsb = (unsigned)(size_t)&tile[l];
      asm volatile("global_load_async_to_lds_b32 %0, %1, %2"
                   :: "v"(ldsb), "v"(goff), "s"(src) : "memory");
#else
      tile[l] = src[goff >> 2];
#endif
    } else {
      tile[l] = 0u;
    }
  }
#if defined(__gfx1250__)
  asm volatile("s_wait_asynccnt 0" ::: "memory");
#endif
  __syncthreads();
#pragma unroll
  for (int k = 0; k < LROWS; ++k) cur[k] = tile[tid + 256 * k];

  for (int it = 0; it < 5; ++it) {
    unsigned nv[LROWS];
#pragma unroll
    for (int k = 0; k < LROWS; ++k) {
      const int l = tid + 256 * k;
      const unsigned m = cur[k];
      unsigned o = m | (m << 1) | (m >> 1);       // in-word horizontal
      if (k > 0)         o |= cur[k - 1];         // row -1 (own register column)
      if (k < LROWS - 1) o |= cur[k + 1];         // row +1
      if (w > 0)         o |= tile[l - 1] >> 31;  // word-boundary horizontal
      if (w < 15)        o |= tile[l + 1] << 31;
      if (b > 0)         o |= tile[l - 16];       // batch -1
      if (b < 15)        o |= tile[l + 16];       // batch +1
      nv[k] = o;
    }
    __syncthreads();
#pragma unroll
    for (int k = 0; k < LROWS; ++k) { tile[tid + 256 * k] = nv[k]; cur[k] = nv[k]; }
    __syncthreads();
  }

  // write back the 16 exact center rows
#pragma unroll
  for (int k = HALO; k < HALO + TROWS; ++k) {
    const int h = h0 - HALO + k;
    const unsigned goff = (unsigned)((b * WPB + h * 16 + w) * 4);
#if defined(__gfx1250__)
    const unsigned ldsb = (unsigned)(size_t)&tile[tid + 256 * k];
    asm volatile("global_store_async_from_lds_b32 %0, %1, %2"
                 :: "v"(goff), "v"(ldsb), "s"(dst) : "memory");
#else
    dst[goff >> 2] = cur[k];
#endif
  }
#if defined(__gfx1250__)
  asm volatile("s_wait_asynccnt 0" ::: "memory");   // s_endpgm also wait-idles
#endif
}

__global__ void k_refine(int* pseu, const float* __restrict__ img,
                         const unsigned* __restrict__ mask, const Scal* __restrict__ sc, int cls) {
  if (!sc->has) return;                              // uniform branch
  const int w = blockIdx.x * 256 + threadIdx.x;
  unsigned bits = mask[w];
  if (!bits) return;
  const float proto = sc->proto;
  const int pbase = w * 32, b = pbase >> 18, r0 = pbase & (HW - 1);
  const float* __restrict__ im1 = img + (size_t)b * CHW_IMG + HW + r0;
  while (bits) {
    const int j = __builtin_ctz(bits);
    bits &= bits - 1;
    const float sim = -logf(fabsf(proto - im1[j]) + 0.01f);
    if (sim > SIM_THR) pseu[pbase + j] = cls;
  }
}

// cross-entropy with ignore + write pseu to output (WMMA reduce)
__global__ void k_loss(const int* __restrict__ pseu, const float* __restrict__ pred,
                       const float* __restrict__ lse, float* __restrict__ outPseu,
                       float* partialS, float* partialC) {
  __shared__ float buf[256];
  float s = 0.f, c = 0.f;
  const int stride = gridDim.x * blockDim.x;
  for (int p = blockIdx.x * blockDim.x + threadIdx.x; p < NPIX; p += stride) {
    const int v = pseu[p];
    outPseu[p] = (float)v;
    if (v != IGNORE_L) {
      const int b = p >> 18, r = p & (HW - 1);
      s += lse[p] - pred[(size_t)b * CHW_PRED + (size_t)v * HW + r];
      c += 1.f;
    }
  }
  const float S = block_reduce_256_wmma(s, buf);
  const float C = block_reduce_256_wmma(c, buf);
  if (threadIdx.x == 0) { partialS[blockIdx.x] = S; partialC[blockIdx.x] = C; }
}

__global__ void k_loss_final(const float* __restrict__ partialS, const float* __restrict__ partialC,
                             float* __restrict__ out, int nparts) {
  __shared__ float buf[256];
  float s = 0.f, c = 0.f;
  for (int i = threadIdx.x; i < nparts; i += 256) { s += partialS[i]; c += partialC[i]; }
  const float S = block_reduce_256_wmma(s, buf);
  const float C = block_reduce_256_wmma(c, buf);
  if (threadIdx.x == 0) out[0] = S / fmaxf(C, 1.0f);
}

// ---------------- launcher ----------------
extern "C" void kernel_launch(void* const* d_in, const int* in_sizes, int n_in,
                              void* d_out, int out_size, void* d_ws, size_t ws_size,
                              hipStream_t stream) {
  (void)in_sizes; (void)n_in; (void)out_size; (void)ws_size;
  const float* img  = (const float*)d_in[0];
  const float* pred = (const float*)d_in[2];
  float* out = (float*)d_out;

  // workspace carve-out (~49.3 MB)
  float*    ent      = (float*)d_ws;
  float*    lse      = ent + NPIX;
  int*      pseu     = (int*)(lse + NPIX);
  unsigned* maskA    = (unsigned*)(pseu + NPIX);
  unsigned* maskB    = maskA + NWORDS;
  unsigned* hist     = maskB + NWORDS;          // 65536 bins
  float*    partialS = (float*)(hist + 65536);  // 1024
  float*    partialC = partialS + 1024;         // 1024
  Scal*     sc       = (Scal*)(partialC + 1024);

  k_softmax<<<NPIX / 256, 256, 0, stream>>>(pred, ent, lse, pseu);

  for (int cls = 1; cls < NUM_CLASS; ++cls) {
    // exact 60th percentile of entropy over {pseu == cls} via 2-level radix select
    k_clear<<<256, 256, 0, stream>>>(hist, sc, 1);
    k_hist_hi<<<2048, 256, 0, stream>>>(pseu, ent, hist, cls);
    k_select_hi<<<1, 256, 0, stream>>>(hist, sc);
    for (int slot = 0; slot < 2; ++slot) {
      k_clear<<<256, 256, 0, stream>>>(hist, sc, 0);
      k_hist_lo<<<2048, 256, 0, stream>>>(pseu, ent, hist, sc, cls, slot);
      k_select_lo<<<1, 256, 0, stream>>>(hist, sc, slot);
    }
    k_thresh<<<1, 32, 0, stream>>>(sc);

    // threshold + mask build + prototype reduction
    k_apply<<<NWORDS / 256, 256, 0, stream>>>(pseu, ent, img, maskA, sc, cls, partialS, partialC);
    k_proto<<<1, 256, 0, stream>>>(partialS, partialC, sc, NWORDS / 256);

    // 10 cross-dilation iterations = 2 launches of the 5-iteration time-tiled kernel
    k_dilate5<<<H_ / TROWS, 256, 0, stream>>>(maskA, maskB);
    k_dilate5<<<H_ / TROWS, 256, 0, stream>>>(maskB, maskA);

    k_refine<<<NWORDS / 256, 256, 0, stream>>>(pseu, img, maskA, sc, cls);
  }

  k_loss<<<1024, 256, 0, stream>>>(pseu, pred, lse, out + 1, partialS, partialC);
  k_loss_final<<<1, 256, 0, stream>>>(partialS, partialC, out, 1024);
}
